// DecoderNet_42640435314832
// MI455X (gfx1250) — matrix operand order repaired and checked by reading, not compiled, not run on hardware
//
#include <hip/hip_runtime.h>
#include <hip/hip_bf16.h>

// ---------------------------------------------------------------------------
// DecoderNet on MI455X (gfx1250).
// Attention MLP is purely linear -> softmax(enc @ u) with
// u = W1[0:H] @ W2 @ W3 @ Wv; ctx and gctx are step-invariant (computed once).
// Per-step: gates GEMM + LSTM cell + logits GEMM [64,512]x[512,30000] with
// v_wmma_f32_16x16x32_bf16.  Each wave owns one N-tile and ALL 4 M-tiles
// (4 accumulators) so B is streamed exactly once per step.  Only the B
// (weight-stream) fragment is prefetched one K-chunk ahead: A comes from an
// L0-resident 64KB buffer, so this hides the L2 latency at +8 VGPRs only
// (the full 5-fragment double-buffer caused scratch spills).
// ---------------------------------------------------------------------------

#define Bsz 64
#define Hd  512
#define WD  1024
#define Vv  30000
#define Ss  256
#define Tm1 27
#define G4H 2048

typedef __bf16 bf16x16 __attribute__((ext_vector_type(16)));
typedef float  f32x8   __attribute__((ext_vector_type(8)));

struct __align__(16) U4 { unsigned int x, y, z, w; };
union Frag { bf16x16 v; U4 q[2]; };

__device__ __forceinline__ unsigned short f2bf(float f) {
    union { float f; unsigned int u; } v; v.f = f;
    unsigned int u = v.u;
    u += 0x7FFFu + ((u >> 16) & 1u);      // round-to-nearest-even
    return (unsigned short)(u >> 16);
}
__device__ __forceinline__ float sigm(float x) { return 1.0f / (1.0f + __expf(-x)); }

// ---- one-time: fp32 [R,C] -> bf16 transposed [C,R] -------------------------
__global__ void k_t_bf16(const float* __restrict__ src, unsigned short* __restrict__ dst,
                         int R, int C, int roff) {
    long long idx = (long long)blockIdx.x * blockDim.x + threadIdx.x;
    if (idx >= (long long)R * C) return;
    int r = (int)(idx / C), c = (int)(idx % C);
    dst[(size_t)c * R + r] = f2bf(src[(size_t)(r + roff) * C + c]);
}

// ---- one-time: u = W1[0:H] @ (W2 @ (W3 @ Wv)) ------------------------------
__global__ void k_u(const float* __restrict__ W1, const float* __restrict__ W2,
                    const float* __restrict__ W3, const float* __restrict__ Wv,
                    float* __restrict__ u) {
    __shared__ float t1[Hd], t2[Hd];
    int i = threadIdx.x;                       // 512 threads
    float a = 0.f;
    for (int j = 0; j < Hd; ++j) a += W3[(size_t)i * Hd + j] * Wv[j];
    t1[i] = a; __syncthreads();
    float b = 0.f;
    for (int j = 0; j < Hd; ++j) b += W2[(size_t)i * Hd + j] * t1[j];
    t2[i] = b; __syncthreads();
    float c = 0.f;
    for (int j = 0; j < Hd; ++j) c += W1[(size_t)i * Hd + j] * t2[j];
    u[i] = c;
}

// ---- one-time: softmax attention + context (step-invariant) ----------------
__global__ void k_attn(const float* __restrict__ enc, const float* __restrict__ u,
                       float* __restrict__ ctx) {
    int b = blockIdx.x, s = threadIdx.x;       // 64 blocks x 256 threads
    const float* eb = enc + (size_t)b * Ss * Hd;
    const float* es = eb + (size_t)s * Hd;
    float sc = 0.f;
    for (int j = 0; j < Hd; ++j) sc += es[j] * u[j];
    __shared__ float aw[Ss];
    __shared__ float red[Ss];
    red[s] = sc; __syncthreads();
    for (int st = 128; st > 0; st >>= 1) {
        if (s < st) red[s] = fmaxf(red[s], red[s + st]);
        __syncthreads();
    }
    float mx = red[0]; __syncthreads();
    float e = __expf(sc - mx);
    aw[s] = e; red[s] = e; __syncthreads();
    for (int st = 128; st > 0; st >>= 1) {
        if (s < st) red[s] += red[s + st];
        __syncthreads();
    }
    float inv = 1.0f / red[0];
    for (int h = s; h < Hd; h += Ss) {
        float acc = 0.f;
        for (int ss = 0; ss < Ss; ++ss) acc += aw[ss] * eb[(size_t)ss * Hd + h];
        ctx[(size_t)b * Hd + h] = acc * inv;
    }
}

// ---- one-time: gctx = ctx @ Wih[WD:WD+H] + bih + bhh -----------------------
__global__ void k_gctx(const float* __restrict__ ctx, const float* __restrict__ Wih,
                       const float* __restrict__ bih, const float* __restrict__ bhh,
                       float* __restrict__ gctx) {
    int idx = blockIdx.x * blockDim.x + threadIdx.x;  // 64*2048
    if (idx >= Bsz * G4H) return;
    int b = idx >> 11, n = idx & (G4H - 1);
    float acc = bih[n] + bhh[n];
    const float* cb = ctx + (size_t)b * Hd;
    for (int j = 0; j < Hd; ++j) acc += cb[j] * Wih[(size_t)(WD + j) * G4H + n];
    gctx[idx] = acc;
}

// ---- one-time: h0->hbf, c=0, ciw=<BOS>=1 -----------------------------------
__global__ void k_init(const float* __restrict__ h0, unsigned short* __restrict__ hbf,
                       float* __restrict__ c, int* __restrict__ ciw) {
    int idx = blockIdx.x * blockDim.x + threadIdx.x;  // 64*512
    if (idx < Bsz * Hd) { hbf[idx] = f2bf(h0[idx]); c[idx] = 0.0f; }
    if (idx < Bsz) ciw[idx] = 1;
}

// ---- per-step: embedding gather -> bf16 ------------------------------------
__global__ void k_embed(const float* __restrict__ emb, const int* __restrict__ ciw,
                        unsigned short* __restrict__ wbf) {
    int b = blockIdx.x;
    int row = ciw[b];
    const float* r = emb + (size_t)row * WD;
    for (int j = threadIdx.x; j < WD; j += blockDim.x)
        wbf[(size_t)b * WD + j] = f2bf(r[j]);
}

// ---- fragment loaders (A row-major [M,K], B pre-transposed [N,K]) ----------
__device__ __forceinline__ bf16x16 loadA(const unsigned short* base, int row, int Kdim,
                                         int k0, int half) {
    Frag f;
    const unsigned short* p = base + (size_t)row * Kdim + k0 + half * 8;
    f.q[0] = *(const U4*)p;            // K = k0+half*8    .. +7
    f.q[1] = *(const U4*)(p + 16);     // K = k0+16+half*8 .. +7
    return f.v;
}
__device__ __forceinline__ bf16x16 loadB(const unsigned short* base, int col, int Kdim,
                                         int k0, int half) {
    Frag f;
    const unsigned short* p = base + (size_t)col * Kdim + k0 + half * 8;
    f.q[0] = *(const U4*)p;            // K = k0+half*8    .. +7
    f.q[1] = *(const U4*)(p + 16);     // K = k0+16+half*8 .. +7
    return f.v;
}

#define WMMA_BF16(A_, B_, C_) \
    __builtin_amdgcn_wmma_f32_16x16x32_bf16(false, A_, false, B_, (short)0, C_, false, false)

// 4xM-tile K-loop with B-only prefetch (one K-chunk ahead).  A fragments are
// loaded in-iteration (L0-resident buffer, short latency); the streamed B
// fragment's L2 latency is hidden behind the 4-WMMA chain.
#define PIPELINED_KLOOP(Abase, Bbase, KDIM)                                          \
    {                                                                                \
        bf16x16 b_ = loadB(Bbase, ncol, KDIM, 0, half);                              \
        _Pragma("clang loop unroll_count(2)")                                        \
        for (int k0 = 0; k0 < (KDIM) - 32; k0 += 32) {                               \
            bf16x16 bn = loadB(Bbase, ncol, KDIM, k0 + 32, half);                    \
            bf16x16 f0 = loadA(Abase, mr,      KDIM, k0, half);                      \
            bf16x16 f1 = loadA(Abase, mr + 16, KDIM, k0, half);                      \
            bf16x16 f2 = loadA(Abase, mr + 32, KDIM, k0, half);                      \
            bf16x16 f3 = loadA(Abase, mr + 48, KDIM, k0, half);                      \
            a0 = WMMA_BF16(f0, b_, a0);                                              \
            a1 = WMMA_BF16(f1, b_, a1);                                              \
            a2 = WMMA_BF16(f2, b_, a2);                                              \
            a3 = WMMA_BF16(f3, b_, a3);                                              \
            b_ = bn;                                                                 \
        }                                                                            \
        {                                                                            \
            int kl = (KDIM) - 32;                                                    \
            bf16x16 f0 = loadA(Abase, mr,      KDIM, kl, half);                      \
            bf16x16 f1 = loadA(Abase, mr + 16, KDIM, kl, half);                      \
            bf16x16 f2 = loadA(Abase, mr + 32, KDIM, kl, half);                      \
            bf16x16 f3 = loadA(Abase, mr + 48, KDIM, kl, half);                      \
            a0 = WMMA_BF16(f0, b_, a0);                                              \
            a1 = WMMA_BF16(f1, b_, a1);                                              \
            a2 = WMMA_BF16(f2, b_, a2);                                              \
            a3 = WMMA_BF16(f3, b_, a3);                                              \
        }                                                                            \
    }

// ---- per-step: gates G = wbf@WihwT^T + hbf@WhhT^T + gctx  (WMMA bf16) ------
__global__ void k_gates(const unsigned short* __restrict__ wbf,    // [64,1024]
                        const unsigned short* __restrict__ hbf,    // [64,512]
                        const unsigned short* __restrict__ WihwT,  // [2048,1024]
                        const unsigned short* __restrict__ WhhT,   // [2048,512]
                        const float* __restrict__ gctx,            // [64,2048]
                        float* __restrict__ G) {                   // [64,2048]
    int wid  = blockIdx.x * (blockDim.x >> 5) + (threadIdx.x >> 5); // 128 waves
    int lane = threadIdx.x & 31;
    int ncol = wid * 16 + (lane & 15);
    int mr   = lane & 15;
    int half = lane >> 4;
    f32x8 a0 = {0.f,0.f,0.f,0.f,0.f,0.f,0.f,0.f};
    f32x8 a1 = a0, a2 = a0, a3 = a0;
    PIPELINED_KLOOP(wbf, WihwT, WD)
    PIPELINED_KLOOP(hbf, WhhT, Hd)
    int mb = half * 8;
    for (int r = 0; r < 8; ++r) {
        size_t o0 = (size_t)(mb + r) * G4H + ncol;
        G[o0] = a0[r] + gctx[o0];
        size_t o1 = o0 + (size_t)16 * G4H;
        G[o1] = a1[r] + gctx[o1];
        size_t o2 = o0 + (size_t)32 * G4H;
        G[o2] = a2[r] + gctx[o2];
        size_t o3 = o0 + (size_t)48 * G4H;
        G[o3] = a3[r] + gctx[o3];
    }
}

// ---- per-step: LSTM cell (i,f,g,o order) -----------------------------------
__global__ void k_lstm(const float* __restrict__ G, float* __restrict__ c,
                       unsigned short* __restrict__ hbf) {
    int idx = blockIdx.x * blockDim.x + threadIdx.x;   // 64*512
    if (idx >= Bsz * Hd) return;
    int b = idx >> 9, n = idx & (Hd - 1);
    const float* g = G + (size_t)b * G4H;
    float i_ = g[n], f_ = g[Hd + n], gg = g[2 * Hd + n], o_ = g[3 * Hd + n];
    float cs = sigm(f_) * c[idx] + sigm(i_) * tanhf(gg);
    float h  = sigm(o_) * tanhf(cs);
    c[idx] = cs;
    hbf[idx] = f2bf(h);
}

// ---- per-step: logits = hbf @ WoutT^T + bout -> d_out[b,t,:]  (WMMA bf16) --
__global__ void k_logits(const unsigned short* __restrict__ hbf,   // [64,512]
                         const unsigned short* __restrict__ WoutT, // [30000,512]
                         const float* __restrict__ bout,
                         float* __restrict__ out_seq, int t) {
    int wid  = blockIdx.x * (blockDim.x >> 5) + (threadIdx.x >> 5);
    if (wid >= Vv / 16) return;                  // wave-uniform exit (EXEC stays full)
    int lane = threadIdx.x & 31;
    int ncol = wid * 16 + (lane & 15);
    int mr   = lane & 15;
    int half = lane >> 4;
    f32x8 a0 = {0.f,0.f,0.f,0.f,0.f,0.f,0.f,0.f};
    f32x8 a1 = a0, a2 = a0, a3 = a0;
    PIPELINED_KLOOP(hbf, WoutT, Hd)
    float bias = bout[ncol];
    int mb = half * 8;
    for (int r = 0; r < 8; ++r) {
        int m = mb + r;
        out_seq[((size_t)(m)      * Tm1 + t) * Vv + ncol] = a0[r] + bias;
        out_seq[((size_t)(m + 16) * Tm1 + t) * Vv + ncol] = a1[r] + bias;
        out_seq[((size_t)(m + 32) * Tm1 + t) * Vv + ncol] = a2[r] + bias;
        out_seq[((size_t)(m + 48) * Tm1 + t) * Vv + ncol] = a3[r] + bias;
    }
}

// ---- per-step: argmax over V, feed back ciw, emit prediction ---------------
__global__ void k_argmax(const float* __restrict__ out_seq, int t,
                         int* __restrict__ ciw, float* __restrict__ preds) {
    int b = blockIdx.x;
    const float* row = out_seq + ((size_t)b * Tm1 + t) * Vv;
    float best = -3.4e38f; int bi = 0;
    for (int j = threadIdx.x; j < Vv; j += blockDim.x) {
        float v = row[j];
        if (v > best) { best = v; bi = j; }
    }
    __shared__ float sv[256];
    __shared__ int   si[256];
    sv[threadIdx.x] = best; si[threadIdx.x] = bi; __syncthreads();
    for (int st = 128; st > 0; st >>= 1) {
        if (threadIdx.x < st) {
            float ov = sv[threadIdx.x + st]; int oi = si[threadIdx.x + st];
            if (ov > sv[threadIdx.x] || (ov == sv[threadIdx.x] && oi < si[threadIdx.x])) {
                sv[threadIdx.x] = ov; si[threadIdx.x] = oi;
            }
        }
        __syncthreads();
    }
    if (threadIdx.x == 0) {
        ciw[b] = si[0];
        preds[(size_t)b * Tm1 + t] = (float)si[0];
    }
}

// ---------------------------------------------------------------------------
extern "C" void kernel_launch(void* const* d_in, const int* in_sizes, int n_in,
                              void* d_out, int out_size, void* d_ws, size_t ws_size,
                              hipStream_t stream) {
    const float* h0   = (const float*)d_in[0];   // [1,64,512]
    const float* enc  = (const float*)d_in[1];   // [64,256,512]
    const float* emb  = (const float*)d_in[4];   // [30000,1024]
    const float* W1   = (const float*)d_in[5];   // [1024,512]
    const float* W2   = (const float*)d_in[7];   // [512,512]
    const float* W3   = (const float*)d_in[9];   // [512,512]
    const float* Wv   = (const float*)d_in[11];  // [512]
    const float* Wih  = (const float*)d_in[12];  // [1536,2048]
    const float* Whh  = (const float*)d_in[13];  // [512,2048]
    const float* bih  = (const float*)d_in[14];  // [2048]
    const float* bhh  = (const float*)d_in[15];  // [2048]
    const float* Wout = (const float*)d_in[16];  // [512,30000]
    const float* bout = (const float*)d_in[17];  // [30000]

    float* out_seq  = (float*)d_out;                       // [64,27,30000]
    float* out_pred = out_seq + (size_t)Bsz * Tm1 * Vv;    // [64,27]

    char* ws = (char*)d_ws;
    unsigned short* WoutT = (unsigned short*)(ws + 0);          // 30,720,000 B
    unsigned short* WihwT = (unsigned short*)(ws + 30720000);   //  4,194,304 B
    unsigned short* WhhT  = (unsigned short*)(ws + 34914304);   //  2,097,152 B
    float*          u     = (float*)         (ws + 37011456);   //      2,048 B
    float*          ctx   = (float*)         (ws + 37013504);   //    131,072 B
    float*          gctx  = (float*)         (ws + 37144576);   //    524,288 B
    float*          Gbuf  = (float*)         (ws + 37668864);   //    524,288 B
    unsigned short* hbf   = (unsigned short*)(ws + 38193152);   //     65,536 B
    float*          cst   = (float*)         (ws + 38258688);   //    131,072 B
    unsigned short* wbf   = (unsigned short*)(ws + 38389760);   //    131,072 B
    int*            ciw   = (int*)           (ws + 38520832);   //        256 B

    // -------- one-time precompute --------
    k_t_bf16<<<(Hd * Vv + 255) / 256, 256, 0, stream>>>(Wout, WoutT, Hd, Vv, 0);
    k_t_bf16<<<(WD * G4H + 255) / 256, 256, 0, stream>>>(Wih, WihwT, WD, G4H, 0);
    k_t_bf16<<<(Hd * G4H + 255) / 256, 256, 0, stream>>>(Whh, WhhT, Hd, G4H, 0);
    k_u<<<1, 512, 0, stream>>>(W1, W2, W3, Wv, u);
    k_attn<<<Bsz, Ss, 0, stream>>>(enc, u, ctx);
    k_gctx<<<(Bsz * G4H + 255) / 256, 256, 0, stream>>>(ctx, Wih, bih, bhh, gctx);
    k_init<<<(Bsz * Hd + 255) / 256, 256, 0, stream>>>(h0, hbf, cst, ciw);

    // -------- serial decode loop (argmax feedback) --------
    const int gate_blocks   = (G4H / 16 + 7) / 8;     // 128 waves -> 16 blocks
    const int logits_blocks = (Vv / 16 + 7) / 8;      // 1875 waves -> 235 blocks
    for (int t = 0; t < Tm1; ++t) {
        k_embed <<<Bsz, 256, 0, stream>>>(emb, ciw, wbf);
        k_gates <<<gate_blocks, 256, 0, stream>>>(wbf, hbf, WihwT, WhhT, gctx, Gbuf);
        k_lstm  <<<(Bsz * Hd + 255) / 256, 256, 0, stream>>>(Gbuf, cst, hbf);
        k_logits<<<logits_blocks, 256, 0, stream>>>(hbf, WoutT, bout, out_seq, t);
        k_argmax<<<Bsz, 256, 0, stream>>>(out_seq, t, ciw, out_pred);
    }
}
